// CrossAttention_5643587027558
// MI455X (gfx1250) — matrix operand order repaired and checked by reading, not compiled, not run on hardware
//
#include <hip/hip_runtime.h>
#include <hip/hip_bf16.h>

typedef __attribute__((ext_vector_type(16))) _Float16 v16h;
typedef __attribute__((ext_vector_type(8)))  float    v8f;

constexpr int Nn = 8192;   // sequence length
constexpr int Cc = 512;    // channels
constexpr int Dk = 64;     // head dim
constexpr int KTILE = 32;  // keys per flash iteration
constexpr int NIT = Nn / KTILE;

// ---------------------------------------------------------------------------
// CDNA5 async global->LDS copy (ASYNCcnt path), with sync fallback.
// Builtin signature (from hipcc diagnostic): (v4i AS1*, v4i AS3*, Imm, Imm).
// ---------------------------------------------------------------------------
#if __has_builtin(__builtin_amdgcn_global_load_async_to_lds_b128) && \
    __has_builtin(__builtin_amdgcn_global_load_async_to_lds_b64)
#define HAVE_ASYNC_LDS 1
#endif

typedef int v4i __attribute__((vector_size(4 * sizeof(int))));
typedef int v2i __attribute__((vector_size(2 * sizeof(int))));
typedef __attribute__((address_space(1))) v4i* g4p;
typedef __attribute__((address_space(3))) v4i* l4p;
typedef __attribute__((address_space(1))) v2i* g2p;
typedef __attribute__((address_space(3))) v2i* l2p;

__device__ __forceinline__ void cp_lds16(void* lds, const void* g) {
#ifdef HAVE_ASYNC_LDS
    __builtin_amdgcn_global_load_async_to_lds_b128(
        (g4p)(uintptr_t)g, (l4p)(unsigned)(uintptr_t)lds, 0, 0);
#else
    *(uint4*)lds = *(const uint4*)g;
#endif
}
__device__ __forceinline__ void cp_lds8(void* lds, const void* g) {
#ifdef HAVE_ASYNC_LDS
    __builtin_amdgcn_global_load_async_to_lds_b64(
        (g2p)(uintptr_t)g, (l2p)(unsigned)(uintptr_t)lds, 0, 0);
#else
    *(uint2*)lds = *(const uint2*)g;
#endif
}

#ifdef HAVE_ASYNC_LDS
# if __has_builtin(__builtin_amdgcn_s_wait_asynccnt)
#  define S_WAIT_ASYNC(n) __builtin_amdgcn_s_wait_asynccnt(n)
# else
#  define S_WAIT_ASYNC(n) asm volatile("s_wait_asynccnt " #n)
# endif
#else
# define S_WAIT_ASYNC(n) do {} while (0)  // sync copies complete before barrier
#endif

__device__ __forceinline__ v8f wmma_f16(v16h a, v16h b, v8f c) {
    return __builtin_amdgcn_wmma_f32_16x16x32_f16(
        false, a, false, b, (short)0, c, false, false);
}

// A fragment (16x32 f16), source row-major src[m*ld + k].
// ISA: lanes 0-15: M=lane, K pairs {0..7,16..23}; lanes 16-31: M=lane-16, K {8..15,24..31}.
__device__ __forceinline__ v16h load_A_rowmajor(const _Float16* src, int ld, int lane) {
    const int m = lane & 15, h = lane >> 4;
    const _Float16* row = src + (size_t)m * ld + (h ? 8 : 0);
    union { v16h v; unsigned u[8]; } f;
#pragma unroll
    for (int i = 0; i < 4; ++i) f.u[i]     = *(const unsigned*)(row + 2 * i);
#pragma unroll
    for (int i = 0; i < 4; ++i) f.u[4 + i] = *(const unsigned*)(row + 16 + 2 * i);
    return f.v;
}

// B fragment (32x16 f16) where B[k][n] = src[n*ld + k].
// ISA: lanes 0-15: N=lane, K {0..7,16..23}; lanes 16-31: N=lane-16, K {8..15,24..31}.
__device__ __forceinline__ v16h load_B_rows(const _Float16* src, size_t ld, int lane) {
    const int n = lane & 15, h = lane >> 4;
    const _Float16* row = src + (size_t)n * ld + (h ? 8 : 0);
    union { v16h v; unsigned u[8]; } f;
#pragma unroll
    for (int i = 0; i < 4; ++i) f.u[i]     = *(const unsigned*)(row + 2 * i);
#pragma unroll
    for (int i = 0; i < 4; ++i) f.u[4 + i] = *(const unsigned*)(row + 16 + 2 * i);
    return f.v;
}

// ---------------------------------------------------------------------------
// Projection GEMM: Y = X(f32,[Nn,Cc]) @ W(f32,[Cc,cout]) -> f16,
// row-major (transposeY=0) or transposed [cout,Nn] (transposeY=1).
// ---------------------------------------------------------------------------
__global__ __launch_bounds__(256) void xattn_proj_kernel(
    const float* __restrict__ X, const float* __restrict__ W,
    _Float16* __restrict__ Y, int cout, int transposeY)
{
    __shared__ _Float16 sX[32][34];
    __shared__ _Float16 sWt[64][34];

    const int qbase = blockIdx.x * 32;
    const int col0  = blockIdx.y * 64;
    const int lane  = threadIdx.x;
    const int wave  = threadIdx.y;
    const int rb    = wave & 1;
    const int cb    = wave >> 1;
    const int tid   = wave * 32 + lane;

    v8f acc = {0.f, 0.f, 0.f, 0.f, 0.f, 0.f, 0.f, 0.f};

    for (int kk = 0; kk < Cc; kk += 32) {
        {
            const int r = tid >> 3, c4 = (tid & 7) * 4;
            const float4 x = *(const float4*)(X + (size_t)(qbase + r) * Cc + kk + c4);
            sX[r][c4 + 0] = (_Float16)x.x; sX[r][c4 + 1] = (_Float16)x.y;
            sX[r][c4 + 2] = (_Float16)x.z; sX[r][c4 + 3] = (_Float16)x.w;
        }
        {
            const int k = tid >> 3, c8 = (tid & 7) * 8;
            const float* wr = W + (size_t)(kk + k) * cout + col0 + c8;
            const float4 w0 = *(const float4*)(wr);
            const float4 w1 = *(const float4*)(wr + 4);
            sWt[c8 + 0][k] = (_Float16)w0.x; sWt[c8 + 1][k] = (_Float16)w0.y;
            sWt[c8 + 2][k] = (_Float16)w0.z; sWt[c8 + 3][k] = (_Float16)w0.w;
            sWt[c8 + 4][k] = (_Float16)w1.x; sWt[c8 + 5][k] = (_Float16)w1.y;
            sWt[c8 + 6][k] = (_Float16)w1.z; sWt[c8 + 7][k] = (_Float16)w1.w;
        }
        __syncthreads();
        v16h a = load_A_rowmajor(&sX[rb * 16][0], 34, lane);
        v16h b = load_B_rows(&sWt[cb * 16][0], 34, lane);
        acc = wmma_f16(a, b, acc);
        __syncthreads();
    }

    const int m = lane & 15, h = lane >> 4;
#pragma unroll
    for (int v = 0; v < 8; ++v) {
        const int row = qbase + rb * 16 + v + 8 * h;
        const int col = col0 + cb * 16 + m;
        if (!transposeY) Y[(size_t)row * cout + col] = (_Float16)acc[v];
        else             Y[(size_t)col * Nn + row]   = (_Float16)acc[v];
    }
}

// ---------------------------------------------------------------------------
// Flash cross-attention: out = softmax(Q K^T) V + resid.
// 512 threads = 16 waves: 4 row-blocks (16 q rows) x 4 col-groups (128 V cols).
// K/V tiles double-buffered in LDS via async global->LDS copies.
// ---------------------------------------------------------------------------
__global__ __launch_bounds__(512) void xattn_flash_kernel(
    const _Float16* __restrict__ Q, const _Float16* __restrict__ Km,
    const _Float16* __restrict__ Vt, const float* __restrict__ resid,
    float* __restrict__ out)
{
    __shared__ _Float16 sV[2][Cc][40];   // [buf][col][key], 80B row stride (16B-aligned)
    __shared__ _Float16 sK[2][KTILE][72];// [buf][key][dk], 144B row stride
    __shared__ _Float16 sP[4][16][34];   // softmax tiles per row-block
    __shared__ float    sScale[4][16];

    const int qbase = blockIdx.x * 64;
    const int lane  = threadIdx.x;
    const int wave  = threadIdx.y;       // 0..15
    const int rb    = wave & 3;          // row block
    const int cg    = wave >> 2;         // col group (128 cols)
    const int m     = lane & 15, h = lane >> 4;
    const int tid   = wave * 32 + lane;  // 0..511

    // stage K/V tiles for iteration `it` into buffer `buf` (5 async ops/thread)
    auto issue_copy = [&](int it, int buf) {
        const int kb = it * KTILE;
#pragma unroll
        for (int t = 0; t < 4; ++t) {
            const int chunk = tid + t * 512;         // 2048 x 16B = 32KB V tile
            const int col   = chunk >> 2;
            const int j8    = (chunk & 3) * 8;
            cp_lds16(&sV[buf][col][j8], Vt + (size_t)col * Nn + kb + j8);
        }
        {
            const int r = tid >> 4, c4 = (tid & 15) * 4; // 512 x 8B = 4KB K tile
            cp_lds8(&sK[buf][r][c4], Km + (size_t)(kb + r) * Dk + c4);
        }
    };

    // Q fragments for this row block (resident)
    const _Float16* qrow = Q + (size_t)(qbase + rb * 16) * Dk;
    const v16h qa0 = load_A_rowmajor(qrow, Dk, lane);
    const v16h qa1 = load_A_rowmajor(qrow + 32, Dk, lane);

    v8f acc[8];
#pragma unroll
    for (int t = 0; t < 8; ++t) acc[t] = (v8f){0.f,0.f,0.f,0.f,0.f,0.f,0.f,0.f};
    float run_max[8], run_sum[8];
#pragma unroll
    for (int v = 0; v < 8; ++v) { run_max[v] = -3.0e38f; run_sum[v] = 0.f; }

    issue_copy(0, 0);
    for (int it = 0; it < NIT; ++it) {
        const int buf = it & 1;
        if (it + 1 < NIT) { issue_copy(it + 1, buf ^ 1); S_WAIT_ASYNC(5); }
        else              { S_WAIT_ASYNC(0); }
        __syncthreads();   // buffer `buf` resident for all waves

        if (cg == 0) {
            // S[16,32] = Q(rb) @ K^T (two 16x16 tiles, Dk=64 in 2 k-steps)
            v16h b00 = load_B_rows(&sK[buf][0][0],   72, lane);
            v16h b01 = load_B_rows(&sK[buf][0][32],  72, lane);
            v16h b10 = load_B_rows(&sK[buf][16][0],  72, lane);
            v16h b11 = load_B_rows(&sK[buf][16][32], 72, lane);
            v8f s0 = {0.f,0.f,0.f,0.f,0.f,0.f,0.f,0.f};
            v8f s1 = {0.f,0.f,0.f,0.f,0.f,0.f,0.f,0.f};
            s0 = wmma_f16(qa0, b00, s0); s0 = wmma_f16(qa1, b01, s0);
            s1 = wmma_f16(qa0, b10, s1); s1 = wmma_f16(qa1, b11, s1);

            // online softmax; C-layout: row = v + 8*h (stats lane-local), col = m
#pragma unroll
            for (int v = 0; v < 8; ++v) {
                float tmax = fmaxf(s0[v], s1[v]);
#pragma unroll
                for (int off = 1; off < 16; off <<= 1)
                    tmax = fmaxf(tmax, __shfl_xor(tmax, off, 16));
                const float nm = fmaxf(run_max[v], tmax);
                const float sc = __expf(run_max[v] - nm);
                run_max[v] = nm;
                const float p0 = __expf(s0[v] - nm);
                const float p1 = __expf(s1[v] - nm);
                float rs = p0 + p1;
#pragma unroll
                for (int off = 1; off < 16; off <<= 1)
                    rs += __shfl_xor(rs, off, 16);
                run_sum[v] = run_sum[v] * sc + rs;
                const int r = v + 8 * h;
                sP[rb][r][m]      = (_Float16)p0;
                sP[rb][r][m + 16] = (_Float16)p1;
                if (m == 0) sScale[rb][r] = sc;
            }
        }
        __syncthreads();   // publish P + scale

        float scv[8];
#pragma unroll
        for (int v = 0; v < 8; ++v) scv[v] = sScale[rb][v + 8 * h];
        const v16h pA = load_A_rowmajor(&sP[rb][0][0], 34, lane);
#pragma unroll
        for (int t = 0; t < 8; ++t) {
            v16h vB = load_B_rows(&sV[buf][cg * 128 + t * 16][0], 40, lane);
#pragma unroll
            for (int v = 0; v < 8; ++v) acc[t][v] *= scv[v];
            acc[t] = wmma_f16(pA, vB, acc[t]);
        }
        __syncthreads();   // reads of `buf` done; next iter may overwrite it
    }

    if (cg == 0) {
#pragma unroll
        for (int v = 0; v < 8; ++v)
            if (m == 0) sScale[rb][v + 8 * h] = 1.0f / run_sum[v];
    }
    __syncthreads();

#pragma unroll
    for (int t = 0; t < 8; ++t) {
#pragma unroll
        for (int v = 0; v < 8; ++v) {
            const int row = qbase + rb * 16 + v + 8 * h;
            const int col = cg * 128 + t * 16 + m;
            const float inv = sScale[rb][v + 8 * h];
            const size_t idx = (size_t)row * Cc + col;
            out[idx] = acc[t][v] * inv + resid[idx];
        }
    }
}

extern "C" void kernel_launch(void* const* d_in, const int* in_sizes, int n_in,
                              void* d_out, int out_size, void* d_ws, size_t ws_size,
                              hipStream_t stream) {
    const float* im1 = (const float*)d_in[0];
    const float* im2 = (const float*)d_in[1];
    const float* Wq  = (const float*)d_in[2];
    const float* Wk  = (const float*)d_in[3];
    const float* Wv  = (const float*)d_in[4];

    _Float16* ws  = (_Float16*)d_ws;
    _Float16* Q1  = ws;
    _Float16* K1  = Q1 + (size_t)Nn * Dk;
    _Float16* Q2  = K1 + (size_t)Nn * Dk;
    _Float16* K2  = Q2 + (size_t)Nn * Dk;
    _Float16* V1t = K2 + (size_t)Nn * Dk;
    _Float16* V2t = V1t + (size_t)Cc * Nn;

    float* out1 = (float*)d_out;
    float* out2 = out1 + (size_t)Nn * Cc;

    const dim3 blk(32, 8);
    const dim3 gqk(Nn / 32, Dk / 64);
    const dim3 gv(Nn / 32, Cc / 64);
    xattn_proj_kernel<<<gqk, blk, 0, stream>>>(im1, Wq, Q1, Dk, 0);
    xattn_proj_kernel<<<gqk, blk, 0, stream>>>(im1, Wk, K1, Dk, 0);
    xattn_proj_kernel<<<gv,  blk, 0, stream>>>(im1, Wv, V1t, Cc, 1);
    xattn_proj_kernel<<<gqk, blk, 0, stream>>>(im2, Wq, Q2, Dk, 0);
    xattn_proj_kernel<<<gqk, blk, 0, stream>>>(im2, Wk, K2, Dk, 0);
    xattn_proj_kernel<<<gv,  blk, 0, stream>>>(im2, Wv, V2t, Cc, 1);

    const dim3 ga(Nn / 64, 1);
    const dim3 ba(32, 16);
    // im1_out = softmax(Q2 K1^T) V1 + im1 ; im2_out = softmax(Q1 K2^T) V2 + im2
    xattn_flash_kernel<<<ga, ba, 0, stream>>>(Q2, K1, V1t, im1, out1);
    xattn_flash_kernel<<<ga, ba, 0, stream>>>(Q1, K2, V2t, im2, out2);
}
